// DCL_11776800325979
// MI455X (gfx1250) — matrix operand order repaired and checked by reading, not compiled, not run on hardware
//
#include <hip/hip_runtime.h>

typedef _Float16 v16h __attribute__((ext_vector_type(16)));
typedef _Float16 v8h  __attribute__((ext_vector_type(8)));
typedef _Float16 v4h  __attribute__((ext_vector_type(4)));
typedef float    v8f  __attribute__((ext_vector_type(8)));

#define NROWS   8192
#define HALF_N  4096
#define DIM     128
#define INV_T   10.0f            // 1 / TEMP
#define LSE_MAX 10.0f            // |cos/TEMP| <= 10 -> fixed logsumexp shift
#define SPLIT   4                // column-sweep split factor
#define NTILES  (NROWS / 16)     // 512 column tiles
#define TPC     (NTILES / SPLIT)
// exp(10c - 10) == exp2(c * 10*log2(e) - 10*log2(e)); fold log2(e) into FMA.
#define EXP2_K  14.4269504088896340736f

#if __has_builtin(__builtin_amdgcn_exp2f)
#define EXP2(x) __builtin_amdgcn_exp2f(x)     // single v_exp_f32
#else
#define EXP2(x) exp2f(x)
#endif

// ---------------------------------------------------------------------------
// Kernel 1: row L2-normalization into f16 workspace (one wave32 per row),
// plus zero-init of the scalar output and the per-row partial buffers.
// ---------------------------------------------------------------------------
__global__ __launch_bounds__(256)
void rownorm_f16_kernel(const float* __restrict__ z1,
                        const float* __restrict__ z2,
                        _Float16* __restrict__ nmat,
                        float* __restrict__ partials,   // rowSum[8192]+rowPos[8192]
                        float* __restrict__ out) {
  const int gid = blockIdx.x * 256 + threadIdx.x;
  if (gid == 0) out[0] = 0.0f;
  if (gid < 2 * NROWS) partials[gid] = 0.0f;

  const int wave = threadIdx.x >> 5;          // 8 waves per block
  const int lane = threadIdx.x & 31;
  const int row  = blockIdx.x * 8 + wave;     // 1024 blocks * 8 = 8192 rows

  const float* src = (row < HALF_N) ? (z1 + (size_t)row * DIM)
                                    : (z2 + (size_t)(row - HALF_N) * DIM);
  float4 v = ((const float4*)src)[lane];
  float ss = v.x * v.x + v.y * v.y + v.z * v.z + v.w * v.w;
#pragma unroll
  for (int t = 16; t >= 1; t >>= 1) ss += __shfl_xor(ss, t, 32);
  const float inv = rsqrtf(fmaxf(ss, 1e-16f));

  v4h h;
  h[0] = (_Float16)(v.x * inv);
  h[1] = (_Float16)(v.y * inv);
  h[2] = (_Float16)(v.z * inv);
  h[3] = (_Float16)(v.w * inv);
  ((v4h*)(nmat + (size_t)row * DIM))[lane] = h;
}

// ---------------------------------------------------------------------------
// Kernel 2: fused Gram tile (WMMA f16->f32) + fixed-shift sum-of-exp.
// grid = (128 row-blocks, 4 column chunks); 4 waves/WG, 16 rows/wave.
// LDS double-buffered: one barrier/iter, global loads overlap WMMA.
// ---------------------------------------------------------------------------
__global__ __launch_bounds__(128)
void ntxent_partial_kernel(const _Float16* __restrict__ nmat,
                           float* __restrict__ rowSum,
                           float* __restrict__ rowPos) {
  __shared__ _Float16 Bs[2][16 * DIM];        // 2 x 4 KB column tiles

  const int tid  = threadIdx.x;
  const int lane = tid & 31;
  const int wave = tid >> 5;
  const int hg   = lane >> 4;                 // half-group: 0 or 1
  const int l15  = lane & 15;
  // Scalarize the wave-uniform row block so the rare-tile test below is a
  // pure SALU compare + s_cbranch (no exec masking in the fast path).
  const int rowBase = __builtin_amdgcn_readfirstlane(blockIdx.x * 64 + wave * 16);
  const int ctBase  = blockIdx.y * TPC;

  // Column tiles that need diag / positive masking for this 16-row block.
  const int diagTile = rowBase >> 4;
  const int posTile  = ((rowBase + HALF_N) & (NROWS - 1)) >> 4;

  // --- A fragments (16 rows x K=128), resident for the whole sweep.
  v16h a[4];
  {
    const _Float16* arow = nmat + (size_t)(rowBase + l15) * DIM;
#pragma unroll
    for (int kk = 0; kk < 4; ++kk) {
      v8h lo = *(const v8h*)(arow + kk * 32 + hg * 16);
      v8h hi = *(const v8h*)(arow + kk * 32 + hg * 16 + 8);
      a[kk] = __builtin_shufflevector(lo, hi,
                0, 1, 2, 3, 4, 5, 6, 7, 8, 9, 10, 11, 12, 13, 14, 15);
    }
  }

  float s[8], pv[8];
#pragma unroll
  for (int r = 0; r < 8; ++r) { s[r] = 0.0f; pv[r] = 0.0f; }

  // --- Prologue: stage first tile into buffer 0.
  uint4 r0, r1;
  {
    const uint4* g = (const uint4*)(nmat + (size_t)ctBase * 16 * DIM);
    r0 = g[tid]; r1 = g[tid + 128];
    uint4* b = (uint4*)(Bs[0]);
    b[tid] = r0; b[tid + 128] = r1;
  }

  for (int t = 0; t < TPC; ++t) {
    const int ct = ctBase + t;

    // Prefetch-to-register for tile t+1 (overlaps WMMA below the barrier).
    if (t + 1 < TPC) {
      const uint4* g = (const uint4*)(nmat + (size_t)(ct + 1) * 16 * DIM);
      r0 = g[tid]; r1 = g[tid + 128];
      __builtin_prefetch(nmat + (size_t)(ct + 2) * 16 * DIM + tid * 16, 0, 3);
    }

    __syncthreads();                          // Bs[t&1] stores visible

    // --- 16x16 tile: 4 chained v_wmma_f32_16x16x32_f16 over K=128.
    v8f c = {};
    const _Float16* brow = Bs[t & 1] + l15 * DIM + hg * 16;
#pragma unroll
    for (int kk = 0; kk < 4; ++kk) {
      v16h b = *(const v16h*)(brow + kk * 32);
      c = __builtin_amdgcn_wmma_f32_16x16x32_f16(
              false, a[kk], false, b, (short)0, c, false, false);
    }

    // --- Epilogue: fixed-shift sum of exp, unconditional accumulate so the
    // rare masked path patches e[] only (no phi copies in the hot path).
    float e[8];
#pragma unroll
    for (int r = 0; r < 8; ++r)               // v_fma + v_exp only
      e[r] = EXP2(fmaf(c[r], EXP2_K, -EXP2_K));

    if (__builtin_expect(ct == diagTile || ct == posTile, 0)) {
      const int j = ct * 16 + l15;
#pragma unroll
      for (int r = 0; r < 8; ++r) {
        const int i = rowBase + hg * 8 + r;
        if (j == ((i + HALF_N) & (NROWS - 1))) {  // positive column
          pv[r] += c[r] * INV_T;              // pos_sim (exactly once per row)
          e[r] = 0.0f;                        // excluded from negatives
        }
        if (j == i) e[r] = 0.0f;              // self mask
      }
    }

#pragma unroll
    for (int r = 0; r < 8; ++r) s[r] += e[r]; // always executed: no phi joins

    // Store staged tile t+1 into the alternate buffer (safe: disjoint buffer;
    // readers of it are past the next barrier).
    if (t + 1 < TPC) {
      uint4* b = (uint4*)(Bs[(t + 1) & 1]);
      b[tid] = r0; b[tid + 128] = r1;
    }
  }

  // --- Combine the 16 lanes sharing each row (plain sums now).
#pragma unroll
  for (int t = 8; t >= 1; t >>= 1) {
#pragma unroll
    for (int r = 0; r < 8; ++r) {
      s[r]  += __shfl_xor(s[r], t, 32);
      pv[r] += __shfl_xor(pv[r], t, 32);
    }
  }

  if (l15 == 0) {
#pragma unroll
    for (int r = 0; r < 8; ++r) {
      const int i = rowBase + hg * 8 + r;
      atomicAdd(&rowSum[i], s[r]);
      atomicAdd(&rowPos[i], pv[r]);
    }
  }
}

// ---------------------------------------------------------------------------
// Kernel 3: per-row loss = LSE_MAX + log(rowSum) - rowPos; mean over rows.
// ---------------------------------------------------------------------------
__global__ __launch_bounds__(256)
void finalize_kernel(const float* __restrict__ rowSum,
                     const float* __restrict__ rowPos,
                     float* __restrict__ out) {
  const int i = blockIdx.x * 256 + threadIdx.x;
  float v = (LSE_MAX + __logf(rowSum[i]) - rowPos[i]) * (1.0f / (float)NROWS);
#pragma unroll
  for (int t = 16; t >= 1; t >>= 1) v += __shfl_xor(v, t, 32);
  if ((threadIdx.x & 31) == 0) atomicAdd(out, v);
}

// ---------------------------------------------------------------------------
extern "C" void kernel_launch(void* const* d_in, const int* in_sizes, int n_in,
                              void* d_out, int out_size, void* d_ws, size_t ws_size,
                              hipStream_t stream) {
  const float* z1 = (const float*)d_in[0];
  const float* z2 = (const float*)d_in[1];
  float* out = (float*)d_out;

  _Float16* nmat = (_Float16*)d_ws;                        // 2 MB
  float* partials = (float*)((char*)d_ws + (size_t)NROWS * DIM * sizeof(_Float16));
  float* rowSum = partials;                                // 32 KB
  float* rowPos = partials + NROWS;                        // 32 KB

  rownorm_f16_kernel<<<NROWS / 8, 256, 0, stream>>>(z1, z2, nmat, partials, out);

  dim3 grid(NROWS / 64, SPLIT);
  ntxent_partial_kernel<<<grid, 128, 0, stream>>>(nmat, rowSum, rowPos);

  finalize_kernel<<<NROWS / 256, 256, 0, stream>>>(rowSum, rowPos, out);
}
